// Flux2ParallelSelfAttention_2250562863383
// MI455X (gfx1250) — matrix operand order repaired and checked by reading, not compiled, not run on hardware
//
#include <hip/hip_runtime.h>
#include <hip/hip_bf16.h>
#include <stdint.h>

// ---------------- types & helpers ----------------

typedef __attribute__((ext_vector_type(16))) __bf16 v16bf;
typedef __attribute__((ext_vector_type(8)))  float  v8f;

union FragA { uint32_t u[8]; v16bf v; };
union FragB { uint32_t u[8]; v16bf v; };

__device__ __forceinline__ uint16_t f2bf(float f) {
    union { float f; uint32_t u; } c; c.f = f;
    uint32_t u = c.u;
    u += 0x7FFFu + ((u >> 16) & 1u);   // round-to-nearest-even
    return (uint16_t)(u >> 16);
}
__device__ __forceinline__ float bf2f(uint16_t h) {
    union { uint32_t u; float f; } c; c.u = ((uint32_t)h) << 16;
    return c.f;
}

// async DMA: global (16B-aligned) -> LDS (16B-aligned), tracked by ASYNCcnt
__device__ __forceinline__ void async_g2l_b128(uint32_t lds_off, const void* gptr) {
    asm volatile("global_load_async_to_lds_b128 %0, %1, off"
                 :: "v"(lds_off), "v"(gptr) : "memory");
}
__device__ __forceinline__ void wait_async0() {
    asm volatile("s_wait_asynccnt 0x0" ::: "memory");
}
__device__ __forceinline__ uint32_t lds_addr32(const void* p) {
    return (uint32_t)(uintptr_t)p;   // low 32 bits of generic ptr = LDS offset
}

// Problem constants
constexpr int cSEQ = 4096;
constexpr int cDIM = 3072;
constexpr int cHEADS = 24;
constexpr int cHD = 128;
constexpr int cINNER = 3072;          // HEADS * HD
constexpr int cMLP = 9216;            // MLP_HIDDEN
constexpr int cPCOLS = 27648;         // 3*INNER + 2*MLP
constexpr int cX2COLS = 12288;        // INNER + MLP

// ---------------- elementwise cast ----------------

__global__ void cast_f32_bf16(const float* __restrict__ in,
                              uint16_t* __restrict__ out, size_t n) {
    size_t i = ((size_t)blockIdx.x * blockDim.x + threadIdx.x) * 4;
    if (i + 3 < n) {
        float4 v = *(const float4*)(in + i);
        out[i + 0] = f2bf(v.x); out[i + 1] = f2bf(v.y);
        out[i + 2] = f2bf(v.z); out[i + 3] = f2bf(v.w);
    } else {
        for (; i < n; ++i) out[i] = f2bf(in[i]);
    }
}

// in: [R][C] f32 row-major  ->  out: [C][R] bf16 row-major
__global__ __launch_bounds__(256) void transpose_cast_bf16(
    const float* __restrict__ in, uint16_t* __restrict__ out, int R, int C) {
    __shared__ float tile[32][33];
    int c0 = blockIdx.x * 32, r0 = blockIdx.y * 32;
    int tx = threadIdx.x, ty = threadIdx.y;         // block (32,8)
#pragma unroll
    for (int j = 0; j < 4; ++j)
        tile[ty + j * 8][tx] = in[(size_t)(r0 + ty + j * 8) * C + c0 + tx];
    __syncthreads();
#pragma unroll
    for (int j = 0; j < 4; ++j)
        out[(size_t)(c0 + ty + j * 8) * R + r0 + tx] = f2bf(tile[tx][ty + j * 8]);
}

// ---------------- bf16 WMMA NT GEMM, async double-buffered ----------------
// C[M][N] = A[M][K] * BT[N][K]^T    (A,BT bf16, K-contiguous; C f32 or bf16)
// Block tile 128x128, 8 waves (4M x 2N), K-step 32.
// Tiles staged via GLOBAL_LOAD_ASYNC_TO_LDS_B128 (ASYNCcnt), double-buffered
// so the DMA for tile i+1 overlaps the WMMAs of tile i.

__global__ __launch_bounds__(256) void gemm_bf16_nt(
    const uint16_t* __restrict__ A, const uint16_t* __restrict__ BT,
    void* __restrict__ Cout, int M, int N, int K, int outIsBf16) {
    __shared__ uint32_t sA[2][128 * 20];   // 128 rows x (16 dw data + 4 pad)
    __shared__ uint32_t sB[2][128 * 20];
    const int t = threadIdx.x;
    const int wave = t >> 5, lane = t & 31;
    const int n16 = lane & 15, half = lane >> 4;
    const int wm = wave & 3, wn = wave >> 2;
    const size_t blockM = (size_t)blockIdx.y * 128;
    const size_t blockN = (size_t)blockIdx.x * 128;

    v8f acc[2][4];
#pragma unroll
    for (int i = 0; i < 2; ++i)
#pragma unroll
        for (int j = 0; j < 4; ++j)
#pragma unroll
            for (int r = 0; r < 8; ++r) acc[i][j][r] = 0.0f;

    const int ldRow = t >> 2;            // 0..63
    const int ldCh  = (t & 3) << 2;      // dword 0,4,8,12

    auto issue_tile = [&](int buf, int k0) {
#pragma unroll
        for (int pass = 0; pass < 2; ++pass) {
            int row = pass * 64 + ldRow;
            async_g2l_b128(lds_addr32(&sA[buf][row * 20 + ldCh]),
                           A + (blockM + row) * (size_t)K + k0 + ldCh * 2);
            async_g2l_b128(lds_addr32(&sB[buf][row * 20 + ldCh]),
                           BT + (blockN + row) * (size_t)K + k0 + ldCh * 2);
        }
    };

    issue_tile(0, 0);
    wait_async0();
    __syncthreads();

    int buf = 0;
    for (int k0 = 0; k0 < K; k0 += 32) {
        if (k0 + 32 < K) issue_tile(buf ^ 1, k0 + 32);   // DMA next tile

        FragA a[2]; FragB b[4];
#pragma unroll
        for (int mi = 0; mi < 2; ++mi) {
            const uint32_t* rp = &sA[buf][(wm * 32 + mi * 16 + n16) * 20];
#pragma unroll
            for (int g = 0; g < 4; ++g) {
                a[mi].u[g]     = rp[g + 4 * half];
                a[mi].u[4 + g] = rp[8 + g + 4 * half];
            }
        }
#pragma unroll
        for (int ni = 0; ni < 4; ++ni) {
            const uint32_t* rp = &sB[buf][(wn * 64 + ni * 16 + n16) * 20];
#pragma unroll
            for (int g = 0; g < 8; ++g) b[ni].u[g] = rp[(g & 3) + 4 * half + 8 * (g >> 2)];
        }
#pragma unroll
        for (int mi = 0; mi < 2; ++mi)
#pragma unroll
            for (int ni = 0; ni < 4; ++ni)
                acc[mi][ni] = __builtin_amdgcn_wmma_f32_16x16x32_bf16(
                    false, a[mi].v, false, b[ni].v, (short)0, acc[mi][ni],
                    false, false);

        wait_async0();        // next tile landed in LDS
        __syncthreads();      // (also waits this wave's ds_loads: safe to overwrite)
        buf ^= 1;
    }

#pragma unroll
    for (int mi = 0; mi < 2; ++mi)
#pragma unroll
        for (int ni = 0; ni < 4; ++ni)
#pragma unroll
            for (int r = 0; r < 8; ++r) {
                size_t gm = blockM + wm * 32 + mi * 16 + r + 8 * half;
                size_t gn = blockN + wn * 64 + ni * 16 + n16;
                float val = acc[mi][ni][r];
                if (outIsBf16) ((uint16_t*)Cout)[gm * N + gn] = f2bf(val);
                else           ((float*)Cout)[gm * N + gn]    = val;
            }
}

// ---------------- QK RMSNorm + RoPE (+ V cast to head-major) ----------------
// grid (SEQ, HEADS), block 128 (one d per thread)

__global__ __launch_bounds__(128) void qk_norm_rope(
    const uint16_t* __restrict__ P, const float* __restrict__ nqw,
    const float* __restrict__ nkw, const float* __restrict__ cosT,
    const float* __restrict__ sinT, uint16_t* __restrict__ Qb,
    uint16_t* __restrict__ Kb, uint16_t* __restrict__ Vb) {
    const int s = blockIdx.x, h = blockIdx.y, d = threadIdx.x;
    const uint16_t* row = P + (size_t)s * cPCOLS;
    float q = bf2f(row[h * cHD + d]);
    float k = bf2f(row[cINNER + h * cHD + d]);
    float v = bf2f(row[2 * cINNER + h * cHD + d]);
    __shared__ float rq[128], rk[128], qs[128], ks[128];
    rq[d] = q * q; rk[d] = k * k;
    __syncthreads();
    for (int off = 64; off > 0; off >>= 1) {
        if (d < off) { rq[d] += rq[d + off]; rk[d] += rk[d + off]; }
        __syncthreads();
    }
    float iq = rsqrtf(rq[0] * (1.0f / 128.0f) + 1e-6f);
    float ik = rsqrtf(rk[0] * (1.0f / 128.0f) + 1e-6f);
    float qn = q * iq * nqw[d];
    float kn = k * ik * nkw[d];
    qs[d] = qn; ks[d] = kn;
    __syncthreads();
    float qrot = (d & 1) ? qs[d - 1] : -qs[d + 1];
    float krot = (d & 1) ? ks[d - 1] : -ks[d + 1];
    float c = cosT[(size_t)s * cHD + d], sn = sinT[(size_t)s * cHD + d];
    size_t o = ((size_t)h * cSEQ + s) * cHD + d;
    Qb[o] = f2bf(qn * c + qrot * sn);
    Kb[o] = f2bf(kn * c + krot * sn);
    Vb[o] = f2bf(v);
}

// ---------------- SiLU(m1)*m2 into X2[:, INNER:] ----------------

__global__ void silu_mul(const uint16_t* __restrict__ P,
                         uint16_t* __restrict__ X2) {
    size_t idx = (size_t)blockIdx.x * blockDim.x + threadIdx.x;
    const size_t total = (size_t)cSEQ * cMLP;
    if (idx >= total) return;
    size_t s = idx / cMLP, j = idx % cMLP;
    float m1 = bf2f(P[s * cPCOLS + 3 * cINNER + j]);
    float m2 = bf2f(P[s * cPCOLS + 3 * cINNER + cMLP + j]);
    float sig = 1.0f / (1.0f + __expf(-m1));
    X2[s * cX2COLS + cINNER + j] = f2bf(m1 * sig * m2);
}

// ---------------- flash attention ----------------
// grid (SEQ/128, HEADS), block 256 (8 waves, 16 q-rows per wave)

__global__ __launch_bounds__(256) void flash_attn(
    const uint16_t* __restrict__ Qb, const uint16_t* __restrict__ Kb,
    const uint16_t* __restrict__ Vb, uint16_t* __restrict__ X2) {
    const int h = blockIdx.y;
    const int qblk = blockIdx.x;
    const int t = threadIdx.x, wave = t >> 5, lane = t & 31;
    const int n16 = lane & 15, half = lane >> 4;
    __shared__ uint32_t sK[32 * 68];      // 32 kv rows x (64 dw d + 4 pad)
    __shared__ uint32_t sV[128 * 20];     // 128 d rows x (16 dw kv + 4 pad)
    __shared__ uint32_t sP[8][16 * 20];   // per-wave probs 16x32 bf16
    const size_t hoff = (size_t)h * cSEQ * cHD;
    const uint16_t* Qp = Qb + hoff;
    const uint16_t* Kp = Kb + hoff;
    const uint16_t* Vp = Vb + hoff;

    // Q fragments for this wave's 16 rows (A-layout, 4 k-steps of 32)
    FragA aQ[4];
    {
        int qrow = qblk * 128 + wave * 16 + n16;
        const uint32_t* qp = (const uint32_t*)(Qp + (size_t)qrow * cHD);
#pragma unroll
        for (int kt = 0; kt < 4; ++kt)
#pragma unroll
            for (int g = 0; g < 4; ++g) {
                aQ[kt].u[g]     = qp[kt * 16 + g + 4 * half];
                aQ[kt].u[4 + g] = qp[kt * 16 + 8 + g + 4 * half];
            }
    }

    v8f acc[8];
#pragma unroll
    for (int ni = 0; ni < 8; ++ni)
#pragma unroll
        for (int r = 0; r < 8; ++r) acc[ni][r] = 0.0f;
    float mrow[8], lrow[8];
#pragma unroll
    for (int r = 0; r < 8; ++r) { mrow[r] = -1e30f; lrow[r] = 0.0f; }
    const float scale = 0.08838834764831845f;   // 1/sqrt(128)

    for (int kv0 = 0; kv0 < cSEQ; kv0 += 32) {
        __syncthreads();
        {   // stage K tile via async DMA (row-major, d contiguous) = B^T for Q*K^T
            int row = t >> 3, dch = (t & 7) * 8;
            const uint16_t* gk = Kp + (size_t)(kv0 + row) * cHD + dch * 2;
            uint32_t lk = lds_addr32(&sK[row * 68 + dch]);
            async_g2l_b128(lk,      gk);
            async_g2l_b128(lk + 16, gk + 8);
        }
        {   // stage V transposed: sV[d][kv] (kv contiguous) = B^T for P*V
            int kv = t >> 3, dbase = (t & 7) * 16;
            const uint16_t* vp = Vp + (size_t)(kv0 + kv) * cHD + dbase;
            uint16_t* sv16 = (uint16_t*)sV;
#pragma unroll
            for (int i = 0; i < 16; ++i)
                sv16[(size_t)(dbase + i) * 40 + kv] = vp[i];
        }
        wait_async0();
        __syncthreads();

        // scores: 16 q rows x 32 kv (two 16x16 n-tiles)
        v8f sc[2];
#pragma unroll
        for (int j = 0; j < 2; ++j) {
#pragma unroll
            for (int r = 0; r < 8; ++r) sc[j][r] = 0.0f;
            const uint32_t* rp = &sK[(j * 16 + n16) * 68];
#pragma unroll
            for (int kt = 0; kt < 4; ++kt) {
                FragB bk;
#pragma unroll
                for (int g = 0; g < 8; ++g) bk.u[g] = rp[kt * 16 + (g & 3) + 4 * half + 8 * (g >> 2)];
                sc[j] = __builtin_amdgcn_wmma_f32_16x16x32_bf16(
                    false, aQ[kt].v, false, bk.v, (short)0, sc[j], false, false);
            }
        }

        // online softmax (row = r + 8*half, row spans 16 adjacent lanes)
        float rowf[8], p0[8], p1[8];
#pragma unroll
        for (int r = 0; r < 8; ++r) {
            float v0 = sc[0][r] * scale, v1 = sc[1][r] * scale;
            float mx = fmaxf(v0, v1);
#pragma unroll
            for (int dd = 1; dd < 16; dd <<= 1)
                mx = fmaxf(mx, __shfl_xor(mx, dd, 32));
            float mnew = fmaxf(mrow[r], mx);
            float f = __expf(mrow[r] - mnew);
            float e0 = __expf(v0 - mnew), e1 = __expf(v1 - mnew);
            float sum = e0 + e1;
#pragma unroll
            for (int dd = 1; dd < 16; dd <<= 1)
                sum += __shfl_xor(sum, dd, 32);
            lrow[r] = lrow[r] * f + sum;
            mrow[r] = mnew; rowf[r] = f;
            p0[r] = e0; p1[r] = e1;
        }
#pragma unroll
        for (int ni = 0; ni < 8; ++ni)
#pragma unroll
            for (int r = 0; r < 8; ++r) acc[ni][r] *= rowf[r];

        // probs -> per-wave LDS, reload as A fragment
        {
            uint16_t* pp = (uint16_t*)&sP[wave][0];
#pragma unroll
            for (int r = 0; r < 8; ++r) {
                int prow = r + 8 * half;
                pp[prow * 40 + n16]      = f2bf(p0[r]);
                pp[prow * 40 + 16 + n16] = f2bf(p1[r]);
            }
        }
        asm volatile("s_wait_dscnt 0" ::: "memory");  // wave-local LDS ordering
        FragA aP;
        {
            const uint32_t* rp = &sP[wave][n16 * 20];
#pragma unroll
            for (int g = 0; g < 4; ++g) {
                aP.u[g]     = rp[g + 4 * half];
                aP.u[4 + g] = rp[8 + g + 4 * half];
            }
        }
        // P * V
#pragma unroll
        for (int ni = 0; ni < 8; ++ni) {
            FragB bv;
            const uint32_t* vp2 = &sV[(ni * 16 + n16) * 20];
#pragma unroll
            for (int g = 0; g < 8; ++g) bv.u[g] = vp2[(g & 3) + 4 * half + 8 * (g >> 2)];
            acc[ni] = __builtin_amdgcn_wmma_f32_16x16x32_bf16(
                false, aP.v, false, bv.v, (short)0, acc[ni], false, false);
        }
    }

    // epilogue: normalize, write into X2[:, 0:INNER]
#pragma unroll
    for (int r = 0; r < 8; ++r) lrow[r] = 1.0f / lrow[r];
#pragma unroll
    for (int ni = 0; ni < 8; ++ni)
#pragma unroll
        for (int r = 0; r < 8; ++r) {
            size_t row = (size_t)qblk * 128 + wave * 16 + r + 8 * half;
            size_t col = (size_t)h * cHD + ni * 16 + n16;
            X2[row * cX2COLS + col] = f2bf(acc[ni][r] * lrow[r]);
        }
}

// ---------------- launcher ----------------

extern "C" void kernel_launch(void* const* d_in, const int* in_sizes, int n_in,
                              void* d_out, int out_size, void* d_ws, size_t ws_size,
                              hipStream_t stream) {
    const float* x      = (const float*)d_in[0];
    const float* w_proj = (const float*)d_in[1];
    const float* w_out  = (const float*)d_in[2];
    const float* nqw    = (const float*)d_in[3];
    const float* nkw    = (const float*)d_in[4];
    const float* cosT   = (const float*)d_in[5];
    const float* sinT   = (const float*)d_in[6];
    float* out = (float*)d_out;

    char* ws = (char*)d_ws;
    size_t off = 0;
    uint16_t* xb  = (uint16_t*)(ws + off); off += (size_t)cSEQ * cDIM * 2;
    uint16_t* wpT = (uint16_t*)(ws + off); off += (size_t)cPCOLS * cDIM * 2;
    uint16_t* woT = (uint16_t*)(ws + off); off += (size_t)cDIM * cX2COLS * 2;
    uint16_t* P   = (uint16_t*)(ws + off); off += (size_t)cSEQ * cPCOLS * 2;
    uint16_t* Qb  = (uint16_t*)(ws + off); off += (size_t)cHEADS * cSEQ * cHD * 2;
    uint16_t* Kb  = (uint16_t*)(ws + off); off += (size_t)cHEADS * cSEQ * cHD * 2;
    uint16_t* Vb  = (uint16_t*)(ws + off); off += (size_t)cHEADS * cSEQ * cHD * 2;
    uint16_t* X2  = (uint16_t*)(ws + off); off += (size_t)cSEQ * cX2COLS * 2;

    // 1) cast x -> bf16
    {
        size_t n = (size_t)cSEQ * cDIM;
        cast_f32_bf16<<<(unsigned)((n / 4 + 255) / 256), 256, 0, stream>>>(x, xb, n);
    }
    // 2) transposed bf16 weights (N-major, K-contiguous)
    transpose_cast_bf16<<<dim3(cPCOLS / 32, cDIM / 32), dim3(32, 8), 0, stream>>>(
        w_proj, wpT, cDIM, cPCOLS);
    transpose_cast_bf16<<<dim3(cDIM / 32, cX2COLS / 32), dim3(32, 8), 0, stream>>>(
        w_out, woT, cX2COLS, cDIM);
    // 3) projection GEMM: P = x @ w_proj   (bf16 out)
    gemm_bf16_nt<<<dim3(cPCOLS / 128, cSEQ / 128), 256, 0, stream>>>(
        xb, wpT, P, cSEQ, cPCOLS, cDIM, 1);
    // 4) qk norm + rope (+ v) -> head-major bf16
    qk_norm_rope<<<dim3(cSEQ, cHEADS), 128, 0, stream>>>(
        P, nqw, nkw, cosT, sinT, Qb, Kb, Vb);
    // 5) gated MLP elementwise
    {
        size_t n = (size_t)cSEQ * cMLP;
        silu_mul<<<(unsigned)((n + 255) / 256), 256, 0, stream>>>(P, X2);
    }
    // 6) flash attention -> X2[:, 0:INNER]
    flash_attn<<<dim3(cSEQ / 128, cHEADS), 256, 0, stream>>>(Qb, Kb, Vb, X2);
    // 7) output GEMM: out = X2 @ w_out   (f32 out)
    gemm_bf16_nt<<<dim3(cDIM / 128, cSEQ / 128), 256, 0, stream>>>(
        X2, woT, out, cSEQ, cDIM, cX2COLS, 0);
}
